// RelationalTransformerUpdate_68753836474572
// MI455X (gfx1250) — matrix operand order repaired and checked by reading, not compiled, not run on hardware
//
#include <hip/hip_runtime.h>

#define B_ 4
#define L_ 512
#define D_ 1024
#define H_ 8
#define NL_ 2
#define NR_ 37
#define NRP_ 64
#define FF_ 4096
#define DH_ 128
#define M_ (B_ * L_)

// Flip to 0 if the gfx1250 assembler rejects the async-LDS mnemonics.
#ifndef USE_ASYNC_COPY
#define USE_ASYNC_COPY 1
#endif

typedef __bf16 bf16;
typedef __attribute__((ext_vector_type(16))) __bf16 v16bf;
typedef __attribute__((ext_vector_type(8))) float v8f;
typedef __attribute__((address_space(3))) bf16 lds_bf16;

union Frag {
  v16bf v;
  bf16 e[16];
  uint4 q[2];
};

__device__ inline v8f wmma_bf16(Frag a, Frag b, v8f c) {
  // D = A(16x32 bf16) * B(32x16 bf16) + C(16x16 f32)
  return __builtin_amdgcn_wmma_f32_16x16x32_bf16(false, a.v, false, b.v,
                                                 (short)0, c, false, false);
}

#if USE_ASYNC_COPY
// 16-byte async global -> LDS copy (GLOBAL_LOAD_ASYNC_TO_LDS_B128,
// tracked by ASYNCcnt; VDST = VGPR holding LDS byte offset, VADDR = 64b addr).
__device__ inline void async_ld16(bf16* lds_dst, const bf16* gsrc) {
  const unsigned loff = (unsigned)(size_t)(lds_bf16*)lds_dst;
  asm volatile("global_load_async_to_lds_b128 %0, %1, off"
               :
               : "v"(loff), "v"(gsrc)
               : "memory");
}
__device__ inline void wait_async0() {
  asm volatile("s_wait_asynccnt 0x0" ::: "memory");
}
#endif

// ---------------------------------------------------------------------------
// Weight repack: f32 [K,N] row-major -> bf16 [K/32][N][32] so that each WMMA
// B-fragment (K 8h..8h+7 and 16+8h..16+8h+7 for one column, h = lane/16) is one pair of b128 loads.
// ---------------------------------------------------------------------------
__global__ __launch_bounds__(256) void pack_w_kernel(const float* __restrict__ W,
                                                     bf16* __restrict__ Wp,
                                                     int K, int N) {
  __shared__ float tile[32][33];
  const int kb = blockIdx.x, nb = blockIdx.y;
  const int t = threadIdx.x;
#pragma unroll
  for (int p = 0; p < 4; ++p) {
    const int r = p * 8 + (t >> 5), c = t & 31;
    tile[r][c] = W[(size_t)(kb * 32 + r) * N + nb * 32 + c];
  }
  __syncthreads();
#pragma unroll
  for (int p = 0; p < 4; ++p) {
    const int n = p * 8 + (t >> 5), kl = t & 31;
    Wp[((size_t)kb * N + nb * 32 + n) * 32 + kl] = (bf16)tile[kl][n];
  }
}

// rel_k f32 [NL][NR][DH] -> bf16 [NL][NRP][DH] zero padded (B-mat for q_rel)
// rel_v f32 [NL][NR][DH] -> bf16 [NL][DH][NRP] transposed (B-mat for attn_rel@rel_v)
__global__ __launch_bounds__(256) void prep_rel_kernel(const float* __restrict__ rk,
                                                       const float* __restrict__ rv,
                                                       bf16* __restrict__ rkp,
                                                       bf16* __restrict__ rvp) {
  const int id = blockIdx.x * 256 + threadIdx.x;  // NL*NRP*DH
  if (id >= NL_ * NRP_ * DH_) return;
  const int d = id & (DH_ - 1);
  const int r = (id >> 7) & (NRP_ - 1);
  const int l = id >> 13;
  float vk = 0.f, vv = 0.f;
  if (r < NR_) {
    vk = rk[((size_t)l * NR_ + r) * DH_ + d];
    vv = rv[((size_t)l * NR_ + r) * DH_ + d];
  }
  rkp[((size_t)l * NRP_ + r) * DH_ + d] = (bf16)vk;
  rvp[((size_t)l * DH_ + d) * NRP_ + r] = (bf16)vv;
}

// q,k,v f32 [B,L,H*DH] -> qb,kb bf16 [B,H,L,DH]; v -> vT bf16 [B,H,DH,L]
__global__ __launch_bounds__(256) void pack_qkv_kernel(const float* __restrict__ q,
                                                       const float* __restrict__ k,
                                                       const float* __restrict__ v,
                                                       bf16* __restrict__ qb,
                                                       bf16* __restrict__ kb,
                                                       bf16* __restrict__ vtb) {
  const size_t id = (size_t)blockIdx.x * 256 + threadIdx.x;  // B*H*L*DH
  const int d = id & (DH_ - 1);
  const int l = (id >> 7) & (L_ - 1);
  const int h = (id >> 16) & (H_ - 1);
  const int b = (int)(id >> 19);
  const size_t src = ((size_t)(b * L_ + l)) * D_ + h * DH_ + d;
  qb[id] = (bf16)q[src];
  kb[id] = (bf16)k[src];
  vtb[(((size_t)(b * H_ + h)) * DH_ + d) * L_ + l] = (bf16)v[src];
}

// ---------------------------------------------------------------------------
// LayerNorm over D=1024 (one row per 256-thread block), bf16 or f32 output.
// ---------------------------------------------------------------------------
template <int F32OUT>
__global__ __launch_bounds__(256) void ln_kernel(const float* __restrict__ x,
                                                 const float* __restrict__ g,
                                                 const float* __restrict__ bta,
                                                 bf16* __restrict__ ob,
                                                 float* __restrict__ of) {
  __shared__ float red[16];
  const int row = blockIdx.x;
  const float* xr = x + (size_t)row * D_;
  const int t = threadIdx.x;
  float vals[4];
  float s = 0.f, ss = 0.f;
#pragma unroll
  for (int i = 0; i < 4; ++i) {
    const float v = xr[t + i * 256];
    vals[i] = v;
    s += v;
    ss += v * v;
  }
#pragma unroll
  for (int off = 16; off; off >>= 1) {
    s += __shfl_down(s, off);
    ss += __shfl_down(ss, off);
  }
  if ((t & 31) == 0) {
    red[t >> 5] = s;
    red[8 + (t >> 5)] = ss;
  }
  __syncthreads();
  s = 0.f;
  ss = 0.f;
#pragma unroll
  for (int i = 0; i < 8; ++i) {
    s += red[i];
    ss += red[8 + i];
  }
  const float mean = s * (1.f / D_);
  const float var = ss * (1.f / D_) - mean * mean;
  const float inv = rsqrtf(var + 1e-5f);
#pragma unroll
  for (int i = 0; i < 4; ++i) {
    const int c = t + i * 256;
    const float y = (vals[i] - mean) * inv * g[c] + bta[c];
    if (F32OUT)
      of[(size_t)row * D_ + c] = y;
    else
      ob[(size_t)row * D_ + c] = (bf16)y;
  }
}

// ---------------------------------------------------------------------------
// Tiled bf16 WMMA GEMM: C[M,N] = A[M,K](bf16,row-major) @ Wp + bias
// Wp pre-packed [K/32][N][32]. Block tile 64x128, 8 waves (4m x 2n),
// each wave 16x64. Double-buffered LDS; fill via async global->LDS copies
// (ASYNCcnt) so the next tile streams in while WMMAs consume the current one.
// ---------------------------------------------------------------------------
template <int RELU, int RES, int OUTF32, int OUTBF16>
__global__ __launch_bounds__(256) void gemm_bf16_kernel(
    const bf16* __restrict__ A, const bf16* __restrict__ Wp,
    const float* __restrict__ bias, const float* __restrict__ resid,
    float* __restrict__ outF, bf16* __restrict__ outB, int M, int N, int K) {
  __shared__ bf16 As[2][64][40];
  __shared__ bf16 Bs[2][128][40];
  const int t = threadIdx.x;
  const int lane = t & 31;
  const int wave = t >> 5;
  const int wm = wave & 3;
  const int wn = wave >> 2;
  const int bm = blockIdx.x, bn = blockIdx.y;

  v8f acc[4] = {};

  const int a_row = t >> 2;
  const int a_col = (t & 3) * 8;
  const int b_col = t >> 1;
  const int b_seg = (t & 1) * 16;

  const bf16* aptr = A + (size_t)(bm * 64 + a_row) * K + a_col;
  const bf16* bptr0 = Wp + ((size_t)(bn * 128 + b_col)) * 32 + b_seg;
  const size_t bstep = (size_t)N * 32;

  const int ar = wm * 16 + (lane & 15);
  const int kh = (lane >> 4) * 8;
  const int kk = (lane >> 4) * 16;

#if USE_ASYNC_COPY
  async_ld16(&As[0][a_row][a_col], aptr);
  async_ld16(&Bs[0][b_col][b_seg], bptr0);
  async_ld16(&Bs[0][b_col][b_seg + 8], bptr0 + 8);
  for (int k0 = 0; k0 < K; k0 += 32) {
    const int cur = (k0 >> 5) & 1;
    wait_async0();     // own async copies for buffer `cur` have landed
    __syncthreads();   // everyone's copies landed; prev reads of `nxt` done
    if (k0 + 32 < K) {
      const int nxt = cur ^ 1;
      const bf16* an = aptr + k0 + 32;
      const bf16* bn = bptr0 + (size_t)((k0 + 32) >> 5) * bstep;
      async_ld16(&As[nxt][a_row][a_col], an);
      async_ld16(&Bs[nxt][b_col][b_seg], bn);
      async_ld16(&Bs[nxt][b_col][b_seg + 8], bn + 8);
    }
    Frag af;
    af.q[0] = *(const uint4*)&As[cur][ar][kh];
    af.q[1] = *(const uint4*)&As[cur][ar][16 + kh];
#pragma unroll
    for (int nt = 0; nt < 4; ++nt) {
      Frag bfr;
      const int bc = wn * 64 + nt * 16 + (lane & 15);
      bfr.q[0] = *(const uint4*)&Bs[cur][bc][kh];
      bfr.q[1] = *(const uint4*)&Bs[cur][bc][16 + kh];
      acc[nt] = wmma_bf16(af, bfr, acc[nt]);
    }
  }
#else
  uint4 ra = *(const uint4*)aptr;
  uint4 rb0 = *(const uint4*)bptr0;
  uint4 rb1 = *(const uint4*)(bptr0 + 8);
  *(uint4*)&As[0][a_row][a_col] = ra;
  *(uint4*)&Bs[0][b_col][b_seg] = rb0;
  *(uint4*)&Bs[0][b_col][b_seg + 8] = rb1;
  for (int k0 = 0; k0 < K; k0 += 32) {
    const int cur = (k0 >> 5) & 1;
    __syncthreads();
    const bool more = (k0 + 32 < K);
    if (more) {
      const bf16* an = aptr + k0 + 32;
      const bf16* bn = bptr0 + (size_t)((k0 + 32) >> 5) * bstep;
      ra = *(const uint4*)an;
      rb0 = *(const uint4*)bn;
      rb1 = *(const uint4*)(bn + 8);
    }
    Frag af;
    af.q[0] = *(const uint4*)&As[cur][ar][kh];
    af.q[1] = *(const uint4*)&As[cur][ar][16 + kh];
#pragma unroll
    for (int nt = 0; nt < 4; ++nt) {
      Frag bfr;
      const int bc = wn * 64 + nt * 16 + (lane & 15);
      bfr.q[0] = *(const uint4*)&Bs[cur][bc][kk];
      bfr.q[1] = *(const uint4*)&Bs[cur][bc][kk + 8];
      acc[nt] = wmma_bf16(af, bfr, acc[nt]);
    }
    if (more) {
      const int nxt = cur ^ 1;
      *(uint4*)&As[nxt][a_row][a_col] = ra;
      *(uint4*)&Bs[nxt][b_col][b_seg] = rb0;
      *(uint4*)&Bs[nxt][b_col][b_seg + 8] = rb1;
    }
  }
#endif

  const int row0 = bm * 64 + wm * 16 + (lane >> 4) * 8;
#pragma unroll
  for (int nt = 0; nt < 4; ++nt) {
    const int col = bn * 128 + wn * 64 + nt * 16 + (lane & 15);
    const float bi = bias[col];
#pragma unroll
    for (int r = 0; r < 8; ++r) {
      const int row = row0 + r;
      float v = acc[nt][r] + bi;
      if (RELU) v = fmaxf(v, 0.f);
      const size_t idx = (size_t)row * N + col;
      if (RES) v += resid[idx];
      if (OUTF32) outF[idx] = v;
      if (OUTBF16) outB[idx] = (bf16)v;
    }
  }
}

// ---------------------------------------------------------------------------
// Fused relation-aware attention. One wave (32 threads) per (b, h, 16-row
// i-tile). Everything in WMMA except softmax + relation gathers.
// ---------------------------------------------------------------------------
__global__ __launch_bounds__(32) void attn_kernel(
    const bf16* __restrict__ qb, const bf16* __restrict__ kb,
    const bf16* __restrict__ vtb, const int* __restrict__ rel,
    const bf16* __restrict__ relk, const bf16* __restrict__ relvT,
    bf16* __restrict__ o) {
  __shared__ float probs[16][520];   // padded stride: conflict-free
  __shared__ float qrel[16][NRP_];   // q . rel_k[r]
  __shared__ float arel[16][NRP_];   // sum_j p[i,j] [rel(i,j)==r]

  const int blk = blockIdx.x;
  const int itile = blk & 31;
  const int h = (blk >> 5) & (H_ - 1);
  const int b = blk >> 8;
  const int i0 = itile * 16;
  const int lane = threadIdx.x;
  const int lo = lane & 15;
  const int hi = lane >> 4;
  const int kh = hi * 8;
  const int kk = hi * 16;

  const bf16* qp = qb + (((size_t)(b * H_ + h)) * L_ + i0) * DH_;
  const bf16* kp = kb + ((size_t)(b * H_ + h)) * L_ * DH_;
  const bf16* vp = vtb + ((size_t)(b * H_ + h)) * DH_ * L_;
  const int* relp = rel + ((size_t)b * L_ + i0) * L_;

  // q A-fragments for all 4 K-steps of DH=128
  Frag qf[4];
#pragma unroll
  for (int ks = 0; ks < 4; ++ks) {
    const bf16* p = qp + (size_t)lo * DH_ + ks * 32 + kk;
    qf[ks].q[0] = *(const uint4*)p;
    qf[ks].q[1] = *(const uint4*)(p + 8);
  }

  for (int idx = lane; idx < 16 * NRP_; idx += 32) ((float*)arel)[idx] = 0.f;

  // q_rel[16, 64] = q @ rel_k^T   (4 N-tiles x 4 K-steps of WMMA)
#pragma unroll
  for (int rt = 0; rt < 4; ++rt) {
    v8f c = {};
#pragma unroll
    for (int ks = 0; ks < 4; ++ks) {
      Frag bf_;
      const bf16* p = relk + ((size_t)(rt * 16 + lo)) * DH_ + ks * 32 + kk;
      bf_.q[0] = *(const uint4*)p;
      bf_.q[1] = *(const uint4*)(p + 8);
      c = wmma_bf16(qf[ks], bf_, c);
    }
#pragma unroll
    for (int r = 0; r < 8; ++r) qrel[hi * 8 + r][rt * 16 + lo] = c[r];
  }
  __syncthreads();

  const float scale = 0.08838834764831843f;  // 1/sqrt(128)

  // scores tile-by-tile: q @ k^T + q_rel[rel] gather
  for (int jt = 0; jt < L_ / 16; ++jt) {
    v8f c = {};
#pragma unroll
    for (int ks = 0; ks < 4; ++ks) {
      Frag bf_;
      const bf16* p = kp + ((size_t)(jt * 16 + lo)) * DH_ + ks * 32 + kk;
      bf_.q[0] = *(const uint4*)p;
      bf_.q[1] = *(const uint4*)(p + 8);
      c = wmma_bf16(qf[ks], bf_, c);
    }
#pragma unroll
    for (int r = 0; r < 8; ++r) {
      const int i = hi * 8 + r;
      const int j = jt * 16 + lo;
      const int rl = relp[(size_t)i * L_ + j];
      probs[i][j] = (c[r] + qrel[i][rl]) * scale;
    }
  }
  __syncthreads();

  // softmax per row (2 lanes per row, combine with shfl_xor 16)
  {
    const int base = hi * 256;
    float m = -3.0e38f;
    for (int c = 0; c < 256; ++c) m = fmaxf(m, probs[lo][base + c]);
    m = fmaxf(m, __shfl_xor(m, 16));
    float sum = 0.f;
    for (int c = 0; c < 256; ++c) {
      const float p = __expf(probs[lo][base + c] - m);
      probs[lo][base + c] = p;
      sum += p;
    }
    sum += __shfl_xor(sum, 16);
    const float inv = 1.f / sum;
    for (int c = 0; c < 256; ++c) {
      const int j = base + c;
      const float p = probs[lo][j] * inv;
      probs[lo][j] = p;
      atomicAdd(&arel[lo][relp[(size_t)lo * L_ + j]], p);  // ds_add_f32
    }
  }
  __syncthreads();

  // o = attn @ v   (16 j-chunks x 8 d-tiles of WMMA)
  v8f oacc[8] = {};
  for (int jc = 0; jc < L_ / 32; ++jc) {
    Frag af;
    const int jb = jc * 32;
#pragma unroll
    for (int i = 0; i < 4; ++i) {
      af.e[2 * i] = (bf16)probs[lo][jb + kh + 2 * i];
      af.e[2 * i + 1] = (bf16)probs[lo][jb + kh + 2 * i + 1];
      af.e[8 + 2 * i] = (bf16)probs[lo][jb + 16 + kh + 2 * i];
      af.e[8 + 2 * i + 1] = (bf16)probs[lo][jb + 16 + kh + 2 * i + 1];
    }
#pragma unroll
    for (int dt = 0; dt < 8; ++dt) {
      Frag vf;
      const bf16* p = vp + ((size_t)(dt * 16 + lo)) * L_ + jb + kh;
      vf.q[0] = *(const uint4*)p;
      vf.q[1] = *(const uint4*)(p + 16);
      oacc[dt] = wmma_bf16(af, vf, oacc[dt]);
    }
  }

  // o += attn_rel @ rel_v   (K = NRP = 64 -> 2 K-steps)
#pragma unroll
  for (int ks = 0; ks < 2; ++ks) {
    Frag af;
    const int rb = ks * 32;
#pragma unroll
    for (int i = 0; i < 4; ++i) {
      af.e[2 * i] = (bf16)arel[lo][rb + kh + 2 * i];
      af.e[2 * i + 1] = (bf16)arel[lo][rb + kh + 2 * i + 1];
      af.e[8 + 2 * i] = (bf16)arel[lo][rb + 16 + kh + 2 * i];
      af.e[8 + 2 * i + 1] = (bf16)arel[lo][rb + 16 + kh + 2 * i + 1];
    }
#pragma unroll
    for (int dt = 0; dt < 8; ++dt) {
      Frag rf;
      const bf16* p = relvT + ((size_t)(dt * 16 + lo)) * NRP_ + rb + kh;
      rf.q[0] = *(const uint4*)p;
      rf.q[1] = *(const uint4*)(p + 16);
      oacc[dt] = wmma_bf16(af, rf, oacc[dt]);
    }
  }

  // write o as bf16 [B, L, D] (heads re-interleaved) for the Wo projection
  bf16* op = o + ((size_t)(b * L_ + i0)) * D_ + h * DH_;
#pragma unroll
  for (int dt = 0; dt < 8; ++dt) {
    const int d = dt * 16 + lo;
#pragma unroll
    for (int r = 0; r < 8; ++r)
      op[(size_t)(hi * 8 + r) * D_ + d] = (bf16)oacc[dt][r];
  }
}

// ---------------------------------------------------------------------------
extern "C" void kernel_launch(void* const* d_in, const int* in_sizes, int n_in,
                              void* d_out, int out_size, void* d_ws,
                              size_t ws_size, hipStream_t stream) {
  (void)in_sizes; (void)n_in; (void)out_size; (void)ws_size;

  const float* x_in = (const float*)d_in[0];
  const int* relations = (const int*)d_in[1];
  const float* Wq = (const float*)d_in[2];
  const float* bq = (const float*)d_in[3];
  const float* Wk = (const float*)d_in[4];
  const float* bk = (const float*)d_in[5];
  const float* Wv = (const float*)d_in[6];
  const float* bv = (const float*)d_in[7];
  const float* Wo = (const float*)d_in[8];
  const float* bo = (const float*)d_in[9];
  const float* rel_k = (const float*)d_in[10];
  const float* rel_v = (const float*)d_in[11];
  const float* W1 = (const float*)d_in[12];
  const float* b1 = (const float*)d_in[13];
  const float* W2 = (const float*)d_in[14];
  const float* b2 = (const float*)d_in[15];
  const float* ln1_g = (const float*)d_in[16];
  const float* ln1_b = (const float*)d_in[17];
  const float* ln2_g = (const float*)d_in[18];
  const float* ln2_b = (const float*)d_in[19];
  const float* lnf_g = (const float*)d_in[20];
  const float* lnf_b = (const float*)d_in[21];

  size_t off = 0;
  auto alloc = [&](size_t bytes) -> void* {
    off = (off + 255) & ~(size_t)255;
    void* p = (char*)d_ws + off;
    off += bytes;
    return p;
  };

  bf16* Wqp = (bf16*)alloc((size_t)NL_ * D_ * D_ * 2);
  bf16* Wkp = (bf16*)alloc((size_t)NL_ * D_ * D_ * 2);
  bf16* Wvp = (bf16*)alloc((size_t)NL_ * D_ * D_ * 2);
  bf16* Wop = (bf16*)alloc((size_t)NL_ * D_ * D_ * 2);
  bf16* W1p = (bf16*)alloc((size_t)NL_ * D_ * FF_ * 2);
  bf16* W2p = (bf16*)alloc((size_t)NL_ * FF_ * D_ * 2);
  bf16* rkp = (bf16*)alloc((size_t)NL_ * NRP_ * DH_ * 2);
  bf16* rvp = (bf16*)alloc((size_t)NL_ * DH_ * NRP_ * 2);
  bf16* hb = (bf16*)alloc((size_t)M_ * D_ * 2);
  bf16* ffb = (bf16*)alloc((size_t)M_ * FF_ * 2);
  bf16* ob = (bf16*)alloc((size_t)M_ * D_ * 2);
  bf16* qbb = (bf16*)alloc((size_t)M_ * D_ * 2);
  bf16* kbb = (bf16*)alloc((size_t)M_ * D_ * 2);
  bf16* vtb = (bf16*)alloc((size_t)M_ * D_ * 2);
  float* xcur = (float*)alloc((size_t)M_ * D_ * 4);
  float* qf = (float*)alloc((size_t)M_ * D_ * 4);
  float* kf = (float*)alloc((size_t)M_ * D_ * 4);
  float* vf = (float*)alloc((size_t)M_ * D_ * 4);

  // one-time per-call weight repack to bf16 WMMA-friendly layout
  for (int l = 0; l < NL_; ++l) {
    pack_w_kernel<<<dim3(D_ / 32, D_ / 32), 256, 0, stream>>>(
        Wq + (size_t)l * D_ * D_, Wqp + (size_t)l * D_ * D_, D_, D_);
    pack_w_kernel<<<dim3(D_ / 32, D_ / 32), 256, 0, stream>>>(
        Wk + (size_t)l * D_ * D_, Wkp + (size_t)l * D_ * D_, D_, D_);
    pack_w_kernel<<<dim3(D_ / 32, D_ / 32), 256, 0, stream>>>(
        Wv + (size_t)l * D_ * D_, Wvp + (size_t)l * D_ * D_, D_, D_);
    pack_w_kernel<<<dim3(D_ / 32, D_ / 32), 256, 0, stream>>>(
        Wo + (size_t)l * D_ * D_, Wop + (size_t)l * D_ * D_, D_, D_);
    pack_w_kernel<<<dim3(D_ / 32, FF_ / 32), 256, 0, stream>>>(
        W1 + (size_t)l * D_ * FF_, W1p + (size_t)l * D_ * FF_, D_, FF_);
    pack_w_kernel<<<dim3(FF_ / 32, D_ / 32), 256, 0, stream>>>(
        W2 + (size_t)l * FF_ * D_, W2p + (size_t)l * FF_ * D_, FF_, D_);
  }
  prep_rel_kernel<<<(NL_ * NRP_ * DH_ + 255) / 256, 256, 0, stream>>>(
      rel_k, rel_v, rkp, rvp);

  hipMemcpyAsync(xcur, x_in, (size_t)M_ * D_ * 4, hipMemcpyDeviceToDevice,
                 stream);

  for (int l = 0; l < NL_; ++l) {
    // --- attention sublayer ---
    ln_kernel<0><<<M_, 256, 0, stream>>>(xcur, ln1_g + l * D_, ln1_b + l * D_,
                                         hb, nullptr);
    gemm_bf16_kernel<0, 0, 1, 0><<<dim3(M_ / 64, D_ / 128), 256, 0, stream>>>(
        hb, Wqp + (size_t)l * D_ * D_, bq + l * D_, nullptr, qf, nullptr, M_,
        D_, D_);
    gemm_bf16_kernel<0, 0, 1, 0><<<dim3(M_ / 64, D_ / 128), 256, 0, stream>>>(
        hb, Wkp + (size_t)l * D_ * D_, bk + l * D_, nullptr, kf, nullptr, M_,
        D_, D_);
    gemm_bf16_kernel<0, 0, 1, 0><<<dim3(M_ / 64, D_ / 128), 256, 0, stream>>>(
        hb, Wvp + (size_t)l * D_ * D_, bv + l * D_, nullptr, vf, nullptr, M_,
        D_, D_);
    pack_qkv_kernel<<<(M_ * D_) / 256, 256, 0, stream>>>(qf, kf, vf, qbb, kbb,
                                                         vtb);
    attn_kernel<<<B_ * H_ * (L_ / 16), 32, 0, stream>>>(
        qbb, kbb, vtb, relations, rkp + (size_t)l * NRP_ * DH_,
        rvp + (size_t)l * DH_ * NRP_, ob);
    gemm_bf16_kernel<0, 1, 1, 0><<<dim3(M_ / 64, D_ / 128), 256, 0, stream>>>(
        ob, Wop + (size_t)l * D_ * D_, bo + l * D_, xcur, xcur, nullptr, M_,
        D_, D_);
    // --- feed-forward sublayer ---
    ln_kernel<0><<<M_, 256, 0, stream>>>(xcur, ln2_g + l * D_, ln2_b + l * D_,
                                         hb, nullptr);
    gemm_bf16_kernel<1, 0, 0, 1><<<dim3(M_ / 64, FF_ / 128), 256, 0, stream>>>(
        hb, W1p + (size_t)l * D_ * FF_, b1 + l * FF_, nullptr, nullptr, ffb,
        M_, FF_, D_);
    gemm_bf16_kernel<0, 1, 1, 0><<<dim3(M_ / 64, D_ / 128), 256, 0, stream>>>(
        ffb, W2p + (size_t)l * FF_ * D_, b2 + l * D_, xcur, xcur, nullptr, M_,
        D_, FF_);
  }

  ln_kernel<1><<<M_, 256, 0, stream>>>(xcur, lnf_g, lnf_b, nullptr,
                                       (float*)d_out);
}
